// NestedCell_59493886984653
// MI455X (gfx1250) — matrix operand order repaired and checked by reading, not compiled, not run on hardware
//
#include <hip/hip_runtime.h>
#include <stdint.h>

#define NNODES 2048
#define BATCH  2
#define FDIM   64
#define HID    128
#define HIN    128
#define NH     4
#define NC     32
#define ROWS   (BATCH * NNODES)   // 4096
#define MASKC  (-1.0e10f)

typedef __attribute__((ext_vector_type(16))) __bf16 v16bf;
typedef __attribute__((ext_vector_type(8)))  float  v8f;

union AB16 { v16bf v; uint4 q[2]; uint16_t h[16]; };

__device__ __forceinline__ uint16_t f2bf(float f) {
  uint32_t u = __float_as_uint(f);
  u += 0x7fffu + ((u >> 16) & 1u);          // round-to-nearest-even
  return (uint16_t)(u >> 16);
}
__device__ __forceinline__ float bf2f(uint16_t h) {
  return __uint_as_float(((uint32_t)h) << 16);
}

// A fragment (16x32 bf16, row-major source, lda in elements).
// Wave32 layout: lane<16 -> row=lane, K=k0..k0+7 & k0+16..k0+23;
//                lane>=16 -> row=lane-16, K=k0+8..k0+15 & k0+24..k0+31.
__device__ __forceinline__ v16bf load_a_g(const uint16_t* A, int lda, int m0, int k0, int lane) {
  AB16 u;
  const uint16_t* p = A + (size_t)(m0 + (lane & 15)) * lda + k0 + ((lane >> 4) << 3);
  u.q[0] = *(const uint4*)(p);
  u.q[1] = *(const uint4*)(p + 16);
  return u.v;
}

// B fragment (32x16 bf16) read from a pre-transposed [N][K] matrix (ldb in elements).
// lane<16 -> col=n0+lane, K=k0..k0+7 & k0+16..k0+23; lane>=16 -> col=n0+lane-16, K=k0+8..k0+15 & k0+24..k0+31.
__device__ __forceinline__ v16bf load_bt_g(const uint16_t* BT, int ldb, int n0, int k0, int lane) {
  AB16 u;
  const uint16_t* p = BT + (size_t)(n0 + (lane & 15)) * ldb + k0 + ((lane >> 4) << 3);
  u.q[0] = *(const uint4*)(p);
  u.q[1] = *(const uint4*)(p + 16);
  return u.v;
}

__device__ __forceinline__ v8f wmma_bf16(v16bf a, v16bf b, v8f c) {
  return __builtin_amdgcn_wmma_f32_16x16x32_bf16(false, a, false, b, (short)0, c, false, false);
}

// ---------------------------------------------------------------- prep kernels
__global__ void k_cvt_bf16(const float* __restrict__ in, uint16_t* __restrict__ out, int n) {
  int i = blockIdx.x * blockDim.x + threadIdx.x;
  if (i < n) out[i] = f2bf(in[i]);
}

// in [K][Nn] row-major f32 -> out [Nn][K] bf16
__global__ void k_transpose_bf16(const float* __restrict__ in, uint16_t* __restrict__ out,
                                 int K, int Nn) {
  int idx = blockIdx.x * blockDim.x + threadIdx.x;
  if (idx >= K * Nn) return;
  int k = idx / Nn, n = idx - k * Nn;
  out[(size_t)n * K + k] = f2bf(in[idx]);
}

// ------------------------------------------------- GAT projection: x @ kernel
// grid (ROWS/16, HIN/16), block 32. Writes row-major bf16 + per-batch transposed copy.
__global__ void k_gat_proj(const uint16_t* __restrict__ xb, const uint16_t* __restrict__ kT,
                           uint16_t* __restrict__ hf_rm, uint16_t* __restrict__ hfT) {
  int mt = blockIdx.x, nt = blockIdx.y, lane = threadIdx.x;
  int m0 = mt << 4, n0 = nt << 4;
  v8f acc = {};
#pragma unroll
  for (int kt = 0; kt < 2; ++kt)
    acc = wmma_bf16(load_a_g(xb, FDIM, m0, kt * 32, lane),
                    load_bt_g(kT, FDIM, n0, kt * 32, lane), acc);
  int rbase = (lane >> 4) << 3, cl = lane & 15;
  int col = n0 + cl;
#pragma unroll
  for (int r = 0; r < 8; ++r) {
    int row = m0 + rbase + r;
    uint16_t vv = f2bf(acc[r]);
    hf_rm[(size_t)row * HIN + col] = vv;
    hfT[(((size_t)(row >> 11)) * HIN + col) * NNODES + (row & (NNODES - 1))] = vv;
  }
}

// --------------------------------------------- per-(node,head) attention dots
__global__ void k_scores(const uint16_t* __restrict__ hf_rm, const float* __restrict__ asf,
                         const float* __restrict__ anb, float* __restrict__ ss,
                         float* __restrict__ sn) {
  int t = blockIdx.x * blockDim.x + threadIdx.x;
  if (t >= ROWS * NH) return;
  int row = t >> 2, h = t & 3;
  const uint16_t* hp = hf_rm + (size_t)row * HIN + h * NC;
  const float* s0 = asf + h * NC;
  const float* s1 = anb + h * NC;
  float a0 = 0.f, a1 = 0.f;
#pragma unroll
  for (int c = 0; c < NC; ++c) {
    float v = bf2f(hp[c]);
    a0 += v * s0[c];
    a1 += v * s1[c];
  }
  ss[t] = a0; sn[t] = a1;
}

// -------------------------------------- softmax row stats (online max + sum)
__global__ void k_stats(const float* __restrict__ a, const float* __restrict__ ss,
                        const float* __restrict__ sn, float* __restrict__ rmax,
                        float* __restrict__ rinvd) {
  __shared__ float sm[128][4];
  __shared__ float sd[128][4];
  int bi = blockIdx.x;            // global row 0..4095 (= b*N+i)
  int tid = threadIdx.x;          // 0..127
  int b = bi >> 11;
  float m[4], d[4], ssv[4];
#pragma unroll
  for (int h = 0; h < 4; ++h) { m[h] = -3.0e38f; d[h] = 0.f; ssv[h] = ss[bi * 4 + h]; }
  const float* arow = a + (size_t)bi * NNODES;
  const float* snb = sn + (((size_t)b) << 11) * 4;
  for (int j = tid; j < NNODES; j += 128) {
    float msk = MASKC * (1.0f - arow[j]);
    float4 s4 = *(const float4*)(snb + j * 4);
    float sv[4] = {s4.x, s4.y, s4.z, s4.w};
#pragma unroll
    for (int h = 0; h < 4; ++h) {
      float s = ssv[h] + sv[h];
      s = (s > 0.f) ? s : 0.2f * s;   // leaky_relu(0.2)
      s += msk;
      if (s > m[h]) { d[h] = d[h] * __expf(m[h] - s) + 1.0f; m[h] = s; }
      else           d[h] += __expf(s - m[h]);
    }
  }
#pragma unroll
  for (int h = 0; h < 4; ++h) { sm[tid][h] = m[h]; sd[tid][h] = d[h]; }
  __syncthreads();
  for (int off = 64; off > 0; off >>= 1) {
    if (tid < off) {
#pragma unroll
      for (int h = 0; h < 4; ++h) {
        float m1 = sm[tid][h], d1 = sd[tid][h];
        float m2 = sm[tid + off][h], d2 = sd[tid + off][h];
        float mm = fmaxf(m1, m2);
        sd[tid][h] = d1 * __expf(m1 - mm) + d2 * __expf(m2 - mm);
        sm[tid][h] = mm;
      }
    }
    __syncthreads();
  }
  if (tid == 0) {
#pragma unroll
    for (int h = 0; h < 4; ++h) {
      rmax[bi * 4 + h] = sm[0][h];
      rinvd[bi * 4 + h] = 1.0f / sd[0][h];
    }
  }
}

// ------------------- fused masked softmax x V aggregation (flash-style, WMMA)
// grid ROWS/16 blocks of one wave; each wave owns a 16-row tile x all 128 cols.
__global__ void k_attn(const float* __restrict__ a, const float* __restrict__ ss,
                       const float* __restrict__ sn, const float* __restrict__ rmax,
                       const float* __restrict__ rinvd, const uint16_t* __restrict__ hfT,
                       const float* __restrict__ gbias, uint16_t* __restrict__ conv_bf) {
  int mt = blockIdx.x, lane = threadIdx.x;
  int m0 = mt << 4;
  int b = m0 >> 11;
  int irow = m0 + (lane & 15);
  float ssv[4], rm[4], rid[4];
#pragma unroll
  for (int h = 0; h < 4; ++h) {
    ssv[h] = ss[irow * 4 + h];
    rm[h]  = rmax[irow * 4 + h];
    rid[h] = rinvd[irow * 4 + h];
  }
  const float* arow = a + (size_t)irow * NNODES;
  const float* snb = sn + (((size_t)b) << 11) * 4;
  const uint16_t* hfb = hfT + (size_t)b * HIN * NNODES;
  v8f acc[8] = {};                       // [head][ntile]
  int koff = (lane >> 4) << 3;           // 0 or 8
  for (int j0 = 0; j0 < NNODES; j0 += 32) {
    int kb = j0 + koff;
    __builtin_prefetch(arow + kb + 256, 0, 0);   // stream the adjacency row
    float4 t0 = *(const float4*)(arow + kb);
    float4 t1 = *(const float4*)(arow + kb + 4);
    float4 t2 = *(const float4*)(arow + kb + 16);
    float4 t3 = *(const float4*)(arow + kb + 20);
    float av[16] = {t0.x, t0.y, t0.z, t0.w, t1.x, t1.y, t1.z, t1.w,
                    t2.x, t2.y, t2.z, t2.w, t3.x, t3.y, t3.z, t3.w};
    AB16 p[4];
#pragma unroll
    for (int e = 0; e < 16; ++e) {
      int j = kb + e + ((e >> 3) << 3);            // halves 8..15 hold k+16
      float msk = MASKC * (1.0f - av[e]);
      float4 s4 = *(const float4*)(snb + j * 4);
      float sv[4] = {s4.x, s4.y, s4.z, s4.w};
#pragma unroll
      for (int h = 0; h < 4; ++h) {
        float s = ssv[h] + sv[h];
        s = (s > 0.f) ? s : 0.2f * s;
        s += msk;
        p[h].h[e] = f2bf(__expf(s - rm[h]) * rid[h]);
      }
    }
#pragma unroll
    for (int h = 0; h < 4; ++h) {
#pragma unroll
      for (int nt = 0; nt < 2; ++nt) {
        int ch = h * 32 + nt * 16 + (lane & 15);
        AB16 vb;
        const uint16_t* vp = hfb + (size_t)ch * NNODES + j0 + ((lane >> 4) << 3);
        vb.q[0] = *(const uint4*)(vp);
        vb.q[1] = *(const uint4*)(vp + 16);
        acc[h * 2 + nt] = wmma_bf16(p[h].v, vb.v, acc[h * 2 + nt]);
      }
    }
  }
  int rbase = (lane >> 4) << 3, cl = lane & 15;
#pragma unroll
  for (int h = 0; h < 4; ++h)
#pragma unroll
    for (int nt = 0; nt < 2; ++nt) {
      int col = h * 32 + nt * 16 + cl;
      float bias = gbias[col];
#pragma unroll
      for (int r = 0; r < 8; ++r) {
        int row = m0 + rbase + r;
        conv_bf[(size_t)row * HIN + col] = f2bf(acc[h * 2 + nt][r] + bias);
      }
    }
}

// ------------------------------------------------------------- fused GRU cell
// One wave per 16-row tile: u,r GEMMs (K=256), r*h through LDS, c GEMM, update.
__global__ void k_gru(const uint16_t* __restrict__ conv_bf, const uint16_t* __restrict__ h_bf,
                      const float* __restrict__ h_prev,
                      const uint16_t* __restrict__ WuT, const uint16_t* __restrict__ WrT,
                      const uint16_t* __restrict__ WcT,
                      const float* __restrict__ bu, const float* __restrict__ br,
                      const float* __restrict__ bc,
                      float* __restrict__ h_out, uint16_t* __restrict__ h_out_bf) {
  __shared__ uint16_t lds_rh[16 * 136];
  int mt = blockIdx.x, lane = threadIdx.x;
  int m0 = mt << 4;
  v8f accU[8] = {}, accR[8] = {};
#pragma unroll
  for (int kt = 0; kt < 8; ++kt) {
    int k0 = kt * 32;
    v16bf af = (kt < 4) ? load_a_g(conv_bf, HIN, m0, k0, lane)
                        : load_a_g(h_bf, HID, m0, k0 - 128, lane);
#pragma unroll
    for (int nt = 0; nt < 8; ++nt) {
      accU[nt] = wmma_bf16(af, load_bt_g(WuT, 256, nt * 16, k0, lane), accU[nt]);
      accR[nt] = wmma_bf16(af, load_bt_g(WrT, 256, nt * 16, k0, lane), accR[nt]);
    }
  }
  int rbase = (lane >> 4) << 3, cl = lane & 15;
#pragma unroll
  for (int nt = 0; nt < 8; ++nt) {
    int col = nt * 16 + cl;
#pragma unroll
    for (int r = 0; r < 8; ++r) {
      int row = m0 + rbase + r;
      int node = row & (NNODES - 1);
      float u = 1.0f / (1.0f + __expf(-(bu[node] + accU[nt][r])));
      accU[nt][r] = u;                                  // keep u for the update
      float rg = 1.0f / (1.0f + __expf(-(br[node] + accR[nt][r])));
      lds_rh[(rbase + r) * 136 + col] = f2bf(rg * h_prev[(size_t)row * HID + col]);
    }
  }
  __syncthreads();
  v8f accC[8] = {};
#pragma unroll
  for (int kt = 0; kt < 8; ++kt) {
    int k0 = kt * 32;
    v16bf af;
    if (kt < 4) {
      af = load_a_g(conv_bf, HIN, m0, k0, lane);
    } else {
      AB16 u;
      const uint16_t* p = &lds_rh[(lane & 15) * 136 + (k0 - 128) + ((lane >> 4) << 3)];
      u.q[0] = *(const uint4*)(p);
      u.q[1] = *(const uint4*)(p + 16);
      af = u.v;
    }
#pragma unroll
    for (int nt = 0; nt < 8; ++nt)
      accC[nt] = wmma_bf16(af, load_bt_g(WcT, 256, nt * 16, k0, lane), accC[nt]);
  }
#pragma unroll
  for (int nt = 0; nt < 8; ++nt) {
    int col = nt * 16 + cl;
#pragma unroll
    for (int r = 0; r < 8; ++r) {
      int row = m0 + rbase + r;
      int node = row & (NNODES - 1);
      float c = tanhf(bc[node] + accC[nt][r]);
      float u = accU[nt][r];
      float hn = u * h_prev[(size_t)row * HID + col] + (1.0f - u) * c;
      h_out[(size_t)row * HID + col] = hn;
      h_out_bf[(size_t)row * HID + col] = f2bf(hn);
    }
  }
}

// ------------------------------------------------------------ hR = h_new @ R
__global__ void k_hr(const uint16_t* __restrict__ hbf, const uint16_t* __restrict__ Rt,
                     uint16_t* __restrict__ hRbf) {
  int mt = blockIdx.x, nt = blockIdx.y, lane = threadIdx.x;
  int m0 = mt << 4, n0 = nt << 4;
  v8f acc = {};
#pragma unroll
  for (int kt = 0; kt < 4; ++kt)
    acc = wmma_bf16(load_a_g(hbf, HID, m0, kt * 32, lane),
                    load_bt_g(Rt, HID, n0, kt * 32, lane), acc);
  int rbase = (lane >> 4) << 3, cl = lane & 15;
#pragma unroll
  for (int r = 0; r < 8; ++r)
    hRbf[(size_t)(m0 + rbase + r) * HID + n0 + cl] = f2bf(acc[r]);
}

// -------------------------------- logits[..,v] = hR @ h^T (interleaved store)
// grid (N/16, N/64, B); one wave computes a 16x64 output tile with K=128.
__global__ void k_bilinear(const uint16_t* __restrict__ hRbf, const uint16_t* __restrict__ hbf,
                           float* __restrict__ out, int v) {
  int it = blockIdx.x, jt = blockIdx.y, b = blockIdx.z;
  int lane = threadIdx.x;
  int i0 = it << 4, j0 = jt << 6;
  const uint16_t* A  = hRbf + (size_t)b * NNODES * HID;
  const uint16_t* BT = hbf  + (size_t)b * NNODES * HID;   // h row-major == (h^T)^T
  v8f acc[4] = {};
#pragma unroll
  for (int kt = 0; kt < 4; ++kt) {
    v16bf af = load_a_g(A, HID, i0, kt * 32, lane);
#pragma unroll
    for (int n = 0; n < 4; ++n)
      acc[n] = wmma_bf16(af, load_bt_g(BT, HID, j0 + n * 16, kt * 32, lane), acc[n]);
  }
  int rbase = (lane >> 4) << 3, cl = lane & 15;
#pragma unroll
  for (int n = 0; n < 4; ++n) {
    int j = j0 + n * 16 + cl;
#pragma unroll
    for (int r = 0; r < 8; ++r) {
      int i = i0 + rbase + r;
      out[(((size_t)b * NNODES + i) * NNODES + j) * 3 + v] = acc[n][r];
    }
  }
}

// ----------------------------------------------------------------------------
extern "C" void kernel_launch(void* const* d_in, const int* in_sizes, int n_in,
                              void* d_out, int out_size, void* d_ws, size_t ws_size,
                              hipStream_t stream) {
  (void)in_sizes; (void)n_in; (void)out_size; (void)ws_size;
  const float* x  = (const float*)d_in[0];
  const float* a  = (const float*)d_in[1];
  const float* hin[3] = {(const float*)d_in[2], (const float*)d_in[3], (const float*)d_in[4]};
  const float* gk  = (const float*)d_in[5];
  const float* gas = (const float*)d_in[6];
  const float* gan = (const float*)d_in[7];
  const float* gb  = (const float*)d_in[8];
  const float* W[3][3]; const float* bb[3][3];
  for (int v = 0; v < 3; ++v)
    for (int g = 0; g < 3; ++g) {
      W[v][g]  = (const float*)d_in[9 + v * 6 + g * 2];
      bb[v][g] = (const float*)d_in[9 + v * 6 + g * 2 + 1];
    }
  const float* R[3] = {(const float*)d_in[27], (const float*)d_in[28], (const float*)d_in[29]};

  char* wp = (char*)d_ws;
  auto carve = [&](size_t bytes) -> char* {
    char* p = wp; wp += (bytes + 255) & ~(size_t)255; return p;
  };
  uint16_t* xb  = (uint16_t*)carve((size_t)ROWS * FDIM * 2);
  uint16_t* gkT = (uint16_t*)carve((size_t)HIN * FDIM * 2);
  uint16_t* WT[3][3];
  for (int v = 0; v < 3; ++v)
    for (int g = 0; g < 3; ++g) WT[v][g] = (uint16_t*)carve((size_t)HID * 256 * 2);
  uint16_t* RT[3];
  for (int v = 0; v < 3; ++v) RT[v] = (uint16_t*)carve((size_t)HID * HID * 2);
  uint16_t* hbf[3];
  for (int v = 0; v < 3; ++v) hbf[v] = (uint16_t*)carve((size_t)ROWS * HID * 2);
  uint16_t* hf_rm = (uint16_t*)carve((size_t)ROWS * HIN * 2);
  uint16_t* hfT   = (uint16_t*)carve((size_t)BATCH * HIN * NNODES * 2);
  float* ss  = (float*)carve((size_t)ROWS * NH * 4);
  float* sn  = (float*)carve((size_t)ROWS * NH * 4);
  float* rmx = (float*)carve((size_t)ROWS * NH * 4);
  float* rid = (float*)carve((size_t)ROWS * NH * 4);
  uint16_t* conv_bf = (uint16_t*)carve((size_t)ROWS * HIN * 2);
  uint16_t* hnew_bf[3];
  for (int v = 0; v < 3; ++v) hnew_bf[v] = (uint16_t*)carve((size_t)ROWS * HID * 2);
  uint16_t* hR_bf[3];
  for (int v = 0; v < 3; ++v) hR_bf[v] = (uint16_t*)carve((size_t)ROWS * HID * 2);

  dim3 b32(32), b128(128), b256(256);

  k_cvt_bf16<<<dim3((ROWS * FDIM + 255) / 256), b256, 0, stream>>>(x, xb, ROWS * FDIM);
  k_transpose_bf16<<<dim3((FDIM * HIN + 255) / 256), b256, 0, stream>>>(gk, gkT, FDIM, HIN);
  for (int v = 0; v < 3; ++v)
    for (int g = 0; g < 3; ++g)
      k_transpose_bf16<<<dim3((256 * HID + 255) / 256), b256, 0, stream>>>(W[v][g], WT[v][g], 256, HID);
  for (int v = 0; v < 3; ++v)
    k_transpose_bf16<<<dim3((HID * HID + 255) / 256), b256, 0, stream>>>(R[v], RT[v], HID, HID);
  for (int v = 0; v < 3; ++v)
    k_cvt_bf16<<<dim3((ROWS * HID + 255) / 256), b256, 0, stream>>>(hin[v], hbf[v], ROWS * HID);

  k_gat_proj<<<dim3(ROWS / 16, HIN / 16), b32, 0, stream>>>(xb, gkT, hf_rm, hfT);
  k_scores<<<dim3((ROWS * NH + 255) / 256), b256, 0, stream>>>(hf_rm, gas, gan, ss, sn);
  k_stats<<<dim3(ROWS), b128, 0, stream>>>(a, ss, sn, rmx, rid);
  k_attn<<<dim3(ROWS / 16), b32, 0, stream>>>(a, ss, sn, rmx, rid, hfT, gb, conv_bf);

  float* out  = (float*)d_out;
  float* hout = out + (size_t)BATCH * NNODES * NNODES * 3;   // hp, hmu, hsg follow logits
  for (int v = 0; v < 3; ++v)
    k_gru<<<dim3(ROWS / 16), b32, 0, stream>>>(conv_bf, hbf[v], hin[v],
        WT[v][0], WT[v][1], WT[v][2], bb[v][0], bb[v][1], bb[v][2],
        hout + (size_t)v * ROWS * HID, hnew_bf[v]);
  for (int v = 0; v < 3; ++v)
    k_hr<<<dim3(ROWS / 16, HID / 16), b32, 0, stream>>>(hnew_bf[v], RT[v], hR_bf[v]);
  for (int v = 0; v < 3; ++v)
    k_bilinear<<<dim3(NNODES / 16, NNODES / 64, BATCH), b32, 0, stream>>>(hR_bf[v], hnew_bf[v], out, v);
}
